// TemporalConvCell_50903952392624
// MI455X (gfx1250) — compile-verified, matrix operand order checked by reading, not run on hardware
//
#include <hip/hip_runtime.h>
#include <hip/hip_bf16.h>

typedef _Float16 v16h __attribute__((ext_vector_type(16)));
typedef float    v8f  __attribute__((ext_vector_type(8)));

#define WPB 8  // waves per block (256 threads, wave32)

__device__ inline v16h vsplat16(_Float16 s) {
  v16h r;
#pragma unroll
  for (int i = 0; i < 16; ++i) r[i] = s;
  return r;
}

// K base for VGPR pair v of a 16-bit 16x32 A fragment (ISA 7.12.2 table)
__device__ inline int a_k0(int v, int hi) {
  return (v < 4 ? 0 : 16) + (hi ? 8 : 0) + ((v & 3) << 1);
}

// Build an A fragment (16x32 f16, M=row = lane%16) from an LDS tile
__device__ inline v16h afrag_lds(const _Float16* tile, int stride, int m, int hi, int koff) {
  const _Float16* row = tile + m * stride + koff;
  v16h a;
#pragma unroll
  for (int v = 0; v < 8; ++v) {
    int k0 = a_k0(v, hi);
    a[2 * v]     = row[k0];
    a[2 * v + 1] = row[k0 + 1];
  }
  return a;
}

// Build an A fragment from a global f32 row (converted to f16)
__device__ inline v16h afrag_g32(const float* row, int hi) {
  v16h a;
#pragma unroll
  for (int v = 0; v < 8; ++v) {
    int k0 = a_k0(v, hi);
    a[2 * v]     = (_Float16)row[k0];
    a[2 * v + 1] = (_Float16)row[k0 + 1];
  }
  return a;
}

// ---------------------------------------------------------------- zero
__global__ void zero_kernel(float* __restrict__ a, int na, float* __restrict__ c, int nc) {
  int i = blockIdx.x * blockDim.x + threadIdx.x;
  int g = gridDim.x * blockDim.x;
  for (int k = i; k < na; k += g) a[k] = 0.0f;
  for (int k = i; k < nc; k += g) c[k] = 0.0f;
}

// ---------------------------------------------------------------- prep:
// swizzle w2 / b2 / root into B-fragment layout; fold TCN+proj into
// three 64x64 matrices M_hg / M_h1 / M_h0 and constant vector c.
__global__ __launch_bounds__(256) void prep_kernel(
    const float* __restrict__ w2, const float* __restrict__ b2,
    const float* __restrict__ root,
    const float* __restrict__ tw1, const float* __restrict__ tb1,
    const float* __restrict__ tw2, const float* __restrict__ tb2,
    const float* __restrict__ tw3, const float* __restrict__ tb3,
    const float* __restrict__ pw,  const float* __restrict__ pb,
    _Float16* __restrict__ wsB,    _Float16* __restrict__ wsBb2,
    _Float16* __restrict__ wsRoot,
    _Float16* __restrict__ wsMhg,  _Float16* __restrict__ wsMh1,
    _Float16* __restrict__ wsMh0,  float* __restrict__ cvec) {
  const int t   = blockIdx.x * blockDim.x + threadIdx.x;
  const int gsz = gridDim.x * blockDim.x;

  // w2 viewed as [1024 K][64 N] -> 128 fragments (32 K-steps x 4 N-tiles)
  for (int i = t; i < 128 * 32 * 16; i += gsz) {
    int slot = i & 15, lane = (i >> 4) & 31, f = i >> 9;
    int v = slot >> 1, h = slot & 1;
    int nn = lane & 15, hih = (lane >> 4) & 1;
    int kl = (v < 4 ? 0 : 16) + (hih ? 8 : 0) + ((v & 3) << 1) + h;
    int j = f >> 2, nt = f & 3;
    wsB[i] = (_Float16)w2[(size_t)(j * 32 + kl) * 64 + nt * 16 + nn];
  }
  // b2 viewed [32][64] -> 4 fragments
  for (int i = t; i < 4 * 32 * 16; i += gsz) {
    int slot = i & 15, lane = (i >> 4) & 31, f = i >> 9;
    int v = slot >> 1, h = slot & 1;
    int nn = lane & 15, hih = (lane >> 4) & 1;
    int kl = (v < 4 ? 0 : 16) + (hih ? 8 : 0) + ((v & 3) << 1) + h;
    wsBb2[i] = (_Float16)b2[(size_t)kl * 64 + f * 16 + nn];
  }
  // root [32][64] -> 4 fragments
  for (int i = t; i < 4 * 32 * 16; i += gsz) {
    int slot = i & 15, lane = (i >> 4) & 31, f = i >> 9;
    int v = slot >> 1, h = slot & 1;
    int nn = lane & 15, hih = (lane >> 4) & 1;
    int kl = (v < 4 ? 0 : 16) + (hih ? 8 : 0) + ((v & 3) << 1) + h;
    wsRoot[i] = (_Float16)root[(size_t)kl * 64 + f * 16 + nn];
  }
  // Folded matrices: M_h1 = A1@P1, M_h0 = A2@P2, M_hg = B1@P1+B2@P2+B3@P3
  for (int i = t; i < 8 * 32 * 16; i += gsz) {
    int slot = i & 15, lane = (i >> 4) & 31, f = i >> 9;
    int v = slot >> 1, h = slot & 1;
    int nn = lane & 15, hih = (lane >> 4) & 1;
    int kl = (v < 4 ? 0 : 16) + (hih ? 8 : 0) + ((v & 3) << 1) + h;
    int ks = f >> 2, nt = f & 3;
    int kk = ks * 32 + kl;          // input channel
    int o  = nt * 16 + nn;          // output channel
    float ahg = 0.f, ah1 = 0.f, ah0 = 0.f;
    for (int q = 0; q < 64; ++q) {
      float p1 = pw[(size_t)q * 64 + o];
      float p2 = pw[(size_t)(64 + q) * 64 + o];
      float p3 = pw[(size_t)(128 + q) * 64 + o];
      ahg += tw1[(size_t)q * 192 + kk * 3 + 1] * p1 +
             tw2[(size_t)q * 192 + kk * 3 + 1] * p2 +
             tw3[(size_t)q * 192 + kk * 3 + 1] * p3;
      ah1 += tw1[(size_t)q * 192 + kk * 3 + 0] * p1;
      ah0 += tw2[(size_t)q * 192 + kk * 3 + 0] * p2;
    }
    wsMhg[i] = (_Float16)ahg;
    wsMh1[i] = (_Float16)ah1;
    wsMh0[i] = (_Float16)ah0;
  }
  // c[o] = concat(tcn biases) @ proj_w + proj_b
  for (int o = t; o < 64; o += gsz) {
    float a = pb[o];
    for (int q = 0; q < 64; ++q)
      a += tb1[q] * pw[(size_t)q * 64 + o] +
           tb2[q] * pw[(size_t)(64 + q) * 64 + o] +
           tb3[q] * pw[(size_t)(128 + q) * 64 + o];
    cvec[o] = a;
  }
}

// ---------------------------------------------------------------- edge kernel:
// per wave: 32 edges (two 16-row M-tiles). h_e = relu(ea@w1+b1);
// msg = Z @ w2v + x_e @ b2v, with the A-fragment for K-chunk j being the
// cached x-fragment scaled by one scalar h_e[m,j] per lane. Each B-fragment
// load now feeds 2 WMMAs (one per M-tile).
__global__ __launch_bounds__(256) void ecc_edge_kernel(
    const float* __restrict__ x, const float* __restrict__ ea,
    const int* __restrict__ src, const int* __restrict__ dst,
    const float* __restrict__ w1, const float* __restrict__ b1,
    const v16h* __restrict__ wsB, const v16h* __restrict__ wsBb2,
    float* __restrict__ agg, float* __restrict__ cnt,
    int E, int nTiles) {
  __shared__ float    s_w1[512];
  __shared__ float    s_b1[32];
  __shared__ float    s_ea[WPB][512];   // 32 edges x 16 attrs
  __shared__ _Float16 s_H[WPB][1024];   // 32 edges x 32
  __shared__ _Float16 s_X[WPB][1024];   // 32 edges x 32

  const int t = threadIdx.x;
  for (int i = t; i < 512; i += 256) s_w1[i] = w1[i];
  if (t < 32) s_b1[t] = b1[t];
  __syncthreads();

  const int w = t >> 5, L = t & 31, m = L & 15, hi = L >> 4;
  const int tile = blockIdx.x * WPB + w;
  if (tile >= nTiles) return;
  const int e0 = tile * 32;
  const int nvalid = min(32, E - e0);

  __builtin_prefetch(dst + e0, 0, 0);

  // stage edge_attr tile [32][16] f32 -> LDS (async CDNA5 path when full)
  if (nvalid == 32) {
    unsigned lds = (unsigned)(size_t)(&s_ea[w][0]) + (unsigned)(L * 16);
    unsigned long long ga =
        (unsigned long long)(const void*)(ea + (size_t)e0 * 16) + (unsigned long long)(L * 16);
    asm volatile(
        "global_load_async_to_lds_b128 %0, %1, off\n\t"
        "global_load_async_to_lds_b128 %0, %1, off offset:512\n\t"
        "global_load_async_to_lds_b128 %0, %1, off offset:1024\n\t"
        "global_load_async_to_lds_b128 %0, %1, off offset:1536"
        :: "v"(lds), "v"(ga) : "memory");
  } else {
    for (int i = L; i < 512; i += 32) {
      int e = i >> 4;
      s_ea[w][i] = (e < nvalid) ? ea[(size_t)(e0 + e) * 16 + (i & 15)] : 0.0f;
    }
  }
  // gather x[src[e]] -> f16 LDS tile [32][32]; one edge per lane
  {
    int sidx = src[min(e0 + L, E - 1)];
    const float4* xr = (const float4*)(x + (size_t)sidx * 32);
    _Float16* xo = s_X[w] + L * 32;
#pragma unroll
    for (int p = 0; p < 8; ++p) {
      float4 v4 = xr[p];
      xo[4 * p + 0] = (_Float16)v4.x;
      xo[4 * p + 1] = (_Float16)v4.y;
      xo[4 * p + 2] = (_Float16)v4.z;
      xo[4 * p + 3] = (_Float16)v4.w;
    }
  }
  asm volatile("s_wait_asynccnt 0" ::: "memory");
  asm volatile("s_wait_dscnt 0" ::: "memory");

  // h_e = relu(ea @ w1 + b1) : lane computes column L for all 32 edges
  {
    float b = s_b1[L];
#pragma unroll
    for (int e = 0; e < 32; ++e) {
      float acc = b;
#pragma unroll
      for (int d = 0; d < 16; ++d) acc += s_ea[w][e * 16 + d] * s_w1[d * 32 + L];
      acc = fmaxf(acc, 0.0f);
      if (e >= nvalid) acc = 0.0f;
      s_H[w][e * 32 + L] = (_Float16)acc;
    }
  }
  asm volatile("s_wait_dscnt 0" ::: "memory");

  const v16h xa0 = afrag_lds(s_X[w], 32, m, hi, 0);            // edges 0..15
  const v16h xa1 = afrag_lds(s_X[w] + 16 * 32, 32, m, hi, 0);  // edges 16..31

  v8f accA[4] = {v8f{}, v8f{}, v8f{}, v8f{}};
  v8f accB[4] = {v8f{}, v8f{}, v8f{}, v8f{}};
#pragma unroll
  for (int j = 0; j < 32; ++j) {
    v16h a0 = xa0 * vsplat16(s_H[w][m * 32 + j]);
    v16h a1 = xa1 * vsplat16(s_H[w][(m + 16) * 32 + j]);
    const v16h* bp = wsB + ((size_t)(j * 4) * 32 + L);
#pragma unroll
    for (int q = 0; q < 4; ++q) {
      v16h bq = bp[(size_t)q * 32];
      accA[q] = __builtin_amdgcn_wmma_f32_16x16x32_f16(false, a0, false, bq, (short)0, accA[q], false, false);
      accB[q] = __builtin_amdgcn_wmma_f32_16x16x32_f16(false, a1, false, bq, (short)0, accB[q], false, false);
    }
  }
  {  // + x_e @ b2v
    const v16h* bp = wsBb2 + L;
#pragma unroll
    for (int q = 0; q < 4; ++q) {
      v16h bq = bp[(size_t)q * 32];
      accA[q] = __builtin_amdgcn_wmma_f32_16x16x32_f16(false, xa0, false, bq, (short)0, accA[q], false, false);
      accB[q] = __builtin_amdgcn_wmma_f32_16x16x32_f16(false, xa1, false, bq, (short)0, accB[q], false, false);
    }
  }

  // scatter-add (C layout: N = lane%16, M = r + 8*hi)
  const int n = L & 15;
#pragma unroll
  for (int r = 0; r < 8; ++r) {
    int mp = r + hi * 8;
    if (mp < nvalid) {
      int de = dst[e0 + mp];
      float* ag = agg + (size_t)de * 64 + n;
      atomicAdd(ag + 0,  accA[0][r]);
      atomicAdd(ag + 16, accA[1][r]);
      atomicAdd(ag + 32, accA[2][r]);
      atomicAdd(ag + 48, accA[3][r]);
    }
    int mp2 = mp + 16;
    if (mp2 < nvalid) {
      int de = dst[e0 + mp2];
      float* ag = agg + (size_t)de * 64 + n;
      atomicAdd(ag + 0,  accB[0][r]);
      atomicAdd(ag + 16, accB[1][r]);
      atomicAdd(ag + 32, accB[2][r]);
      atomicAdd(ag + 48, accB[3][r]);
    }
  }
  if (L < nvalid) atomicAdd(&cnt[dst[e0 + L]], 1.0f);
}

// ---------------------------------------------------------------- node kernel:
// h_graph = relu(agg/cnt + x@root + bias);
// h_out = h_graph@M_hg + h_prev[:,2]@M_h1 + h_prev[:,1]@M_h0 + c; emit h_hist.
__global__ __launch_bounds__(256) void node_kernel(
    const float* __restrict__ x, const float* __restrict__ h_prev,
    const float* __restrict__ bias,
    const v16h* __restrict__ wsRoot,
    const v16h* __restrict__ wsMhg, const v16h* __restrict__ wsMh1,
    const v16h* __restrict__ wsMh0, const float* __restrict__ cvec,
    const float* __restrict__ cnt,
    float* __restrict__ out_h /* in: agg, out: h_out */,
    float* __restrict__ out_hist,
    int N, int nTiles) {
  __shared__ _Float16 s_A[WPB][1024];

  const int t = threadIdx.x;
  const int w = t >> 5, L = t & 31, m = L & 15, hi = L >> 4, n = L & 15;
  const int tile = blockIdx.x * WPB + w;
  if (tile >= nTiles) return;
  const int n0 = tile * 16;
  const int nvalid = min(16, N - n0);

  // stage x tile [16][32] f16
  {
    int node = min(n0 + m, N - 1);
    const float* xr = x + (size_t)node * 32 + hi * 16;
    _Float16* xo = s_A[w] + m * 32 + hi * 16;
#pragma unroll
    for (int k = 0; k < 16; ++k) xo[k] = (_Float16)xr[k];
  }
  asm volatile("s_wait_dscnt 0" ::: "memory");
  const v16h xa = afrag_lds(s_A[w], 32, m, hi, 0);

  // x @ root (K=32 -> one WMMA per N-tile)
  v8f xr0 = {}, xr1 = {}, xr2 = {}, xr3 = {};
  xr0 = __builtin_amdgcn_wmma_f32_16x16x32_f16(false, xa, false, wsRoot[0 * 32 + L], (short)0, xr0, false, false);
  xr1 = __builtin_amdgcn_wmma_f32_16x16x32_f16(false, xa, false, wsRoot[1 * 32 + L], (short)0, xr1, false, false);
  xr2 = __builtin_amdgcn_wmma_f32_16x16x32_f16(false, xa, false, wsRoot[2 * 32 + L], (short)0, xr2, false, false);
  xr3 = __builtin_amdgcn_wmma_f32_16x16x32_f16(false, xa, false, wsRoot[3 * 32 + L], (short)0, xr3, false, false);

  // h_graph = relu(mean + x@root + bias)
  float bo[4];
#pragma unroll
  for (int q = 0; q < 4; ++q) bo[q] = bias[q * 16 + n];
  float hg[4][8];
#pragma unroll
  for (int r = 0; r < 8; ++r) {
    int mp = r + hi * 8;
    int node = min(n0 + mp, N - 1);
    float c = fmaxf(cnt[node], 1.0f);
    const float* ag = out_h + (size_t)node * 64 + n;
    hg[0][r] = fmaxf(ag[0]  / c + xr0[r] + bo[0], 0.0f);
    hg[1][r] = fmaxf(ag[16] / c + xr1[r] + bo[1], 0.0f);
    hg[2][r] = fmaxf(ag[32] / c + xr2[r] + bo[2], 0.0f);
    hg[3][r] = fmaxf(ag[48] / c + xr3[r] + bo[3], 0.0f);
  }
  // stash h_graph f16 tile [16][64] for A-fragment building
#pragma unroll
  for (int r = 0; r < 8; ++r) {
    int mp = r + hi * 8;
    _Float16* hr = s_A[w] + mp * 64 + n;
    hr[0]  = (_Float16)hg[0][r];
    hr[16] = (_Float16)hg[1][r];
    hr[32] = (_Float16)hg[2][r];
    hr[48] = (_Float16)hg[3][r];
  }
  asm volatile("s_wait_dscnt 0" ::: "memory");

  // accumulators start at folded constant c
  v8f fa[4];
#pragma unroll
  for (int q = 0; q < 4; ++q) {
    float cv = cvec[q * 16 + n];
#pragma unroll
    for (int r = 0; r < 8; ++r) fa[q][r] = cv;
  }
  // + h_graph @ M_hg  (K=64 -> 2 k-steps)
#pragma unroll
  for (int ks = 0; ks < 2; ++ks) {
    v16h a = afrag_lds(s_A[w], 64, m, hi, ks * 32);
#pragma unroll
    for (int q = 0; q < 4; ++q)
      fa[q] = __builtin_amdgcn_wmma_f32_16x16x32_f16(false, a, false, wsMhg[(size_t)(ks * 4 + q) * 32 + L], (short)0, fa[q], false, false);
  }
  // + h_prev[:,2,:] @ M_h1
#pragma unroll
  for (int ks = 0; ks < 2; ++ks) {
    int node = min(n0 + m, N - 1);
    v16h a = afrag_g32(h_prev + ((size_t)node * 3 + 2) * 64 + ks * 32, hi);
#pragma unroll
    for (int q = 0; q < 4; ++q)
      fa[q] = __builtin_amdgcn_wmma_f32_16x16x32_f16(false, a, false, wsMh1[(size_t)(ks * 4 + q) * 32 + L], (short)0, fa[q], false, false);
  }
  // + h_prev[:,1,:] @ M_h0
#pragma unroll
  for (int ks = 0; ks < 2; ++ks) {
    int node = min(n0 + m, N - 1);
    v16h a = afrag_g32(h_prev + ((size_t)node * 3 + 1) * 64 + ks * 32, hi);
#pragma unroll
    for (int q = 0; q < 4; ++q)
      fa[q] = __builtin_amdgcn_wmma_f32_16x16x32_f16(false, a, false, wsMh0[(size_t)(ks * 4 + q) * 32 + L], (short)0, fa[q], false, false);
  }

  // write h_out (overwrites agg region; each row owned by exactly one wave)
#pragma unroll
  for (int r = 0; r < 8; ++r) {
    int mp = r + hi * 8;
    if (mp < nvalid) {
      float* o = out_h + (size_t)(n0 + mp) * 64 + n;
      o[0]  = fa[0][r];
      o[16] = fa[1][r];
      o[32] = fa[2][r];
      o[48] = fa[3][r];
    }
  }
  // h_hist slots 0,1 = h_prev slots 1,2 (contiguous 128 floats at +64)
  for (int idx = L; idx < 2048; idx += 32) {
    int node = idx >> 7, rem = idx & 127;
    if (node < nvalid)
      out_hist[(size_t)(n0 + node) * 192 + rem] = h_prev[(size_t)(n0 + node) * 192 + 64 + rem];
  }
  // h_hist slot 2 = h_graph (f32 from registers)
#pragma unroll
  for (int r = 0; r < 8; ++r) {
    int mp = r + hi * 8;
    if (mp < nvalid) {
      float* o = out_hist + (size_t)(n0 + mp) * 192 + 128 + n;
      o[0]  = hg[0][r];
      o[16] = hg[1][r];
      o[32] = hg[2][r];
      o[48] = hg[3][r];
    }
  }
}

// ---------------------------------------------------------------- launch
extern "C" void kernel_launch(void* const* d_in, const int* in_sizes, int n_in,
                              void* d_out, int out_size, void* d_ws, size_t ws_size,
                              hipStream_t stream) {
  const float* x      = (const float*)d_in[0];
  const float* ea     = (const float*)d_in[1];
  const float* h_prev = (const float*)d_in[2];
  const int*   ei     = (const int*)d_in[3];
  const float* w1     = (const float*)d_in[4];
  const float* b1     = (const float*)d_in[5];
  const float* w2     = (const float*)d_in[6];
  const float* b2     = (const float*)d_in[7];
  const float* root   = (const float*)d_in[8];
  const float* bias   = (const float*)d_in[9];
  const float* tw1    = (const float*)d_in[10];
  const float* tb1    = (const float*)d_in[11];
  const float* tw2    = (const float*)d_in[12];
  const float* tb2    = (const float*)d_in[13];
  const float* tw3    = (const float*)d_in[14];
  const float* tb3    = (const float*)d_in[15];
  const float* pw     = (const float*)d_in[16];
  const float* pb     = (const float*)d_in[17];

  const int N = in_sizes[0] / 32;   // IN_CH = 32
  const int E = in_sizes[1] / 16;   // EDGE_DIM = 16
  const int* srcp = ei;
  const int* dstp = ei + E;

  float* out_h    = (float*)d_out;
  float* out_hist = (float*)d_out + (size_t)N * 64;

  char* ws = (char*)d_ws;
  size_t off = 0;
  float* cnt = (float*)(ws + off);          off += (((size_t)N * 4) + 63) & ~(size_t)63;
  _Float16* wsB    = (_Float16*)(ws + off); off += (size_t)128 * 32 * 16 * 2;  // 131072 B
  _Float16* wsBb2  = (_Float16*)(ws + off); off += (size_t)4 * 32 * 16 * 2;
  _Float16* wsRoot = (_Float16*)(ws + off); off += (size_t)4 * 32 * 16 * 2;
  _Float16* wsMhg  = (_Float16*)(ws + off); off += (size_t)8 * 32 * 16 * 2;
  _Float16* wsMh1  = (_Float16*)(ws + off); off += (size_t)8 * 32 * 16 * 2;
  _Float16* wsMh0  = (_Float16*)(ws + off); off += (size_t)8 * 32 * 16 * 2;
  float* cvec      = (float*)(ws + off);    off += 64 * 4;

  // 1) zero agg (in d_out) + cnt each launch (deterministic under replay)
  zero_kernel<<<512, 256, 0, stream>>>(out_h, N * 64, cnt, N);
  // 2) weight swizzle + TCN/proj folding
  prep_kernel<<<64, 256, 0, stream>>>(w2, b2, root, tw1, tb1, tw2, tb2, tw3, tb3,
                                      pw, pb, wsB, wsBb2, wsRoot, wsMhg, wsMh1, wsMh0, cvec);
  // 3) edge GEMM + scatter (32 edges per wave)
  int nTilesE = (E + 31) / 32;
  int ebl = (nTilesE + WPB - 1) / WPB;
  ecc_edge_kernel<<<ebl, 256, 0, stream>>>(x, ea, srcp, dstp, w1, b1,
                                           (const v16h*)wsB, (const v16h*)wsBb2,
                                           out_h, cnt, E, nTilesE);
  // 4) node side: mean + root + relu, folded TCN+proj, hist
  int nTilesN = (N + 15) / 16;
  int nbl = (nTilesN + WPB - 1) / WPB;
  node_kernel<<<nbl, 256, 0, stream>>>(x, h_prev, bias,
                                       (const v16h*)wsRoot, (const v16h*)wsMhg,
                                       (const v16h*)wsMh1, (const v16h*)wsMh0,
                                       cvec, cnt, out_h, out_hist, N, nTilesN);
}
